// Head_Graph_63823214019225
// MI455X (gfx1250) — matrix operand order repaired and checked by reading, not compiled, not run on hardware
//
#include <hip/hip_runtime.h>

#define BATCH 4
#define NNODE 4096
#define HDIM  128
#define FDIM  128
#define WAVES 4           // waves per workgroup (128 threads)
#define BN    32          // column block (nodes per j-iteration)
#define NEGBIG (-1.0e30f)

typedef __attribute__((ext_vector_type(16))) __bf16 v16bf;
typedef __attribute__((ext_vector_type(8)))  float  v8f;

union U4BF8 { uint4 u; __bf16 h[8]; };
union F4U   { float4 f4; float f[4]; };

// load 16 contiguous bf16 (32B) into a v16bf via two b128 loads
__device__ __forceinline__ v16bf load16bf(const __bf16* p) {
    U4BF8 a, b;
    a.u = *(const uint4*)p;
    b.u = *(const uint4*)(p + 8);
    v16bf r;
#pragma unroll
    for (int i = 0; i < 8; ++i) { r[i] = a.h[i]; r[i + 8] = b.h[i]; }
    return r;
}

// load A-frag halves: 8 bf16 at p, 8 bf16 at p+16 (split-K A layout)
__device__ __forceinline__ v16bf load16bf_split(const __bf16* p) {
    U4BF8 a, b;
    a.u = *(const uint4*)p;
    b.u = *(const uint4*)(p + 16);
    v16bf r;
#pragma unroll
    for (int i = 0; i < 8; ++i) { r[i] = a.h[i]; r[i + 8] = b.h[i]; }
    return r;
}

// ---------------------------------------------------------------------------
// Kernel 1: q = x Wq^T, k = x Wk^T (bf16, [B*N, F]); v = x Wv^T stored
// transposed as vt[B][F][N] (bf16) so P@V B-fragments are contiguous.
// ---------------------------------------------------------------------------
__global__ __launch_bounds__(WAVES * 32) void proj_kernel(
    const float* __restrict__ x,  const float* __restrict__ Wq,
    const float* __restrict__ Wk, const float* __restrict__ Wv,
    __bf16* __restrict__ qb, __bf16* __restrict__ kb, __bf16* __restrict__ vt)
{
    const int tid  = threadIdx.x;
    const int wave = tid >> 5, lane = tid & 31;
    const int half = lane >> 4, lo = lane & 15;
    const long rowbase = (long)(blockIdx.x * WAVES + wave) * 16;  // row in [B*N]

    // A fragments of x: 16 rows x 128 K, bf16.  lanes 0-15: M=lane, K in
    // {0..15}; lanes 16-31: same M, K in {16..31}  (per 32-K chunk)
    v16bf xa[4];
    {
        const float* xr = x + (rowbase + lo) * HDIM;
#pragma unroll
        for (int kc = 0; kc < 4; ++kc) {
            const float* p0 = xr + kc * 32 + 16 * half;
            v16bf a;
#pragma unroll
            for (int i = 0; i < 8; ++i) {
                a[i]     = (__bf16)p0[i];
                a[i + 8] = (__bf16)p0[8 + i];
            }
            xa[kc] = a;
        }
    }

    const float* Ws[3] = { Wq, Wk, Wv };
    const int b        = (int)(rowbase / NNODE);
    const int nodebase = (int)(rowbase % NNODE);

#pragma unroll
    for (int w = 0; w < 3; ++w) {
#pragma unroll
        for (int ft = 0; ft < 8; ++ft) {
            v8f c = {};
#pragma unroll
            for (int kc = 0; kc < 4; ++kc) {
                // B-frag: B[k=h][n=f] = W[f][h]; lane: f = ft*16+lo,
                // h = kc*32 + 16*half + (0..15) -> 16 contiguous floats
                const float* wp = Ws[w] + (ft * 16 + lo) * HDIM + kc * 32 + 16 * half;
                v16bf bf;
#pragma unroll
                for (int i = 0; i < 16; ++i) bf[i] = (__bf16)wp[i];
                c = __builtin_amdgcn_wmma_f32_16x16x32_bf16(
                        false, xa[kc], false, bf, (short)0, c, false, false);
            }
            if (w < 2) {
                __bf16* dst = (w == 0) ? qb : kb;
#pragma unroll
                for (int r = 0; r < 8; ++r) {
                    const int m = r + 8 * half;
                    dst[(rowbase + m) * FDIM + ft * 16 + lo] = (__bf16)c[r];
                }
            } else {
#pragma unroll
                for (int r = 0; r < 8; ++r) {
                    const int m = r + 8 * half;
                    const int f = ft * 16 + lo;
                    vt[((long)b * FDIM + f) * NNODE + nodebase + m] = (__bf16)c[r];
                }
            }
        }
    }
}

// ---------------------------------------------------------------------------
// Kernel 2: fused flash-style attention.  Each wave owns a 16-row strip and
// all 128 output features.  out = softmax(mask(QK^T*s)) @ V  +  adj @ V.
// adj (256 MB) is streamed exactly once.
// ---------------------------------------------------------------------------
__global__ __launch_bounds__(WAVES * 32) void attn_kernel(
    const float*  __restrict__ adj, const __bf16* __restrict__ qb,
    const __bf16* __restrict__ kb,  const __bf16* __restrict__ vt,
    float* __restrict__ out)
{
    __shared__ float ldsP[WAVES][16 * 36];   // P tile  [16][32], padded stride 36
    __shared__ float ldsA[WAVES][16 * 36];   // adj tile

    const int tid  = threadIdx.x;
    const int wave = tid >> 5, lane = tid & 31;
    const int half = lane >> 4, lo = lane & 15;
    const int b       = blockIdx.y;
    const int rowbase = (blockIdx.x * WAVES + wave) * 16;
    const float scale = 0.08838834764831845f;   // 128^-0.5

    // Q A-fragments (persist across the whole j loop)
    v16bf qa[4];
    {
        const __bf16* qr = qb + (long)(b * NNODE + rowbase + lo) * FDIM;
#pragma unroll
        for (int kc = 0; kc < 4; ++kc)
            qa[kc] = load16bf(qr + kc * 32 + 16 * half);
    }

    v8f accS[8], accA[8];
    float mrun[8], lrun[8];
#pragma unroll
    for (int ft = 0; ft < 8; ++ft) {
#pragma unroll
        for (int r = 0; r < 8; ++r) { accS[ft][r] = 0.0f; accA[ft][r] = 0.0f; }
    }
#pragma unroll
    for (int r = 0; r < 8; ++r) { mrun[r] = NEGBIG; lrun[r] = 0.0f; }

    const float* adjrow = adj + ((long)b * NNODE + rowbase) * NNODE;

    for (int jb = 0; jb < NNODE; jb += BN) {
        // prefetch next adj tile (HBM-bound stream)
        if (jb + BN < NNODE) {
#pragma unroll
            for (int r = 0; r < 8; ++r)
                __builtin_prefetch(adjrow + (long)(r + 8 * half) * NNODE + jb + BN + lo * 2, 0, 0);
        }

        // S = Q @ K^T : two 16x16 node tiles, K-dim = 128 features
        v8f s0 = {}, s1 = {};
#pragma unroll
        for (int kc = 0; kc < 4; ++kc) {
            // B-frag: B[k=feature][n=node]; lane: node = jb + t*16 + lo,
            // features kc*32 + 16*half + (0..15) contiguous in kb row
            const __bf16* kp0 = kb + (long)(b * NNODE + jb + lo) * FDIM + kc * 32 + 16 * half;
            s0 = __builtin_amdgcn_wmma_f32_16x16x32_bf16(
                     false, qa[kc], false, load16bf(kp0), (short)0, s0, false, false);
            s1 = __builtin_amdgcn_wmma_f32_16x16x32_bf16(
                     false, qa[kc], false, load16bf(kp0 + 16 * FDIM), (short)0, s1, false, false);
        }

        // mask, online softmax, stage P & adj into LDS for layout transpose
#pragma unroll
        for (int r = 0; r < 8; ++r) {
            const int m = r + 8 * half;
            const float* ar = adjrow + (long)m * NNODE + jb;
            const float a0 = ar[lo];
            const float a1 = ar[16 + lo];
            const float w0 = (a0 != 0.0f) ? s0[r] * scale : NEGBIG;
            const float w1 = (a1 != 0.0f) ? s1[r] * scale : NEGBIG;

            float mx = fmaxf(w0, w1);
            mx = fmaxf(mx, __shfl_xor(mx, 1, 32));
            mx = fmaxf(mx, __shfl_xor(mx, 2, 32));
            mx = fmaxf(mx, __shfl_xor(mx, 4, 32));
            mx = fmaxf(mx, __shfl_xor(mx, 8, 32));

            const float mn = fmaxf(mrun[r], mx);
            const float rf = __expf(mrun[r] - mn);
            const float e0 = (a0 != 0.0f) ? __expf(w0 - mn) : 0.0f;
            const float e1 = (a1 != 0.0f) ? __expf(w1 - mn) : 0.0f;

            float sm = e0 + e1;
            sm += __shfl_xor(sm, 1, 32);
            sm += __shfl_xor(sm, 2, 32);
            sm += __shfl_xor(sm, 4, 32);
            sm += __shfl_xor(sm, 8, 32);

            lrun[r] = lrun[r] * rf + sm;
            mrun[r] = mn;
#pragma unroll
            for (int ft = 0; ft < 8; ++ft) accS[ft][r] *= rf;

            ldsP[wave][m * 36 + lo]      = e0;
            ldsP[wave][m * 36 + 16 + lo] = e1;
            ldsA[wave][m * 36 + lo]      = a0;
            ldsA[wave][m * 36 + 16 + lo] = a1;
        }
        // same-wave DS ops are in-order: no barrier needed (wave-private tiles)

        // Read P and adj back in A-fragment layout (row = lo, split-K cols)
        v16bf pfrag, afrag;
        {
            const float* Pm = &ldsP[wave][lo * 36];
            const float* Am = &ldsA[wave][lo * 36];
            F4U c0, c1, c2, c3;
            c0.f4 = *(const float4*)&Pm[8 * half];
            c1.f4 = *(const float4*)&Pm[8 * half + 4];
            c2.f4 = *(const float4*)&Pm[8 * half + 16];
            c3.f4 = *(const float4*)&Pm[8 * half + 20];
#pragma unroll
            for (int i = 0; i < 4; ++i) {
                pfrag[i]      = (__bf16)c0.f[i];
                pfrag[4 + i]  = (__bf16)c1.f[i];
                pfrag[8 + i]  = (__bf16)c2.f[i];
                pfrag[12 + i] = (__bf16)c3.f[i];
            }
            c0.f4 = *(const float4*)&Am[8 * half];
            c1.f4 = *(const float4*)&Am[8 * half + 4];
            c2.f4 = *(const float4*)&Am[8 * half + 16];
            c3.f4 = *(const float4*)&Am[8 * half + 20];
#pragma unroll
            for (int i = 0; i < 4; ++i) {
                afrag[i]      = (__bf16)c0.f[i];
                afrag[4 + i]  = (__bf16)c1.f[i];
                afrag[8 + i]  = (__bf16)c2.f[i];
                afrag[12 + i] = (__bf16)c3.f[i];
            }
        }

        // accS += P @ V ; accA += adj @ V   (V from transposed vt[B][F][N])
#pragma unroll
        for (int ft = 0; ft < 8; ++ft) {
            const __bf16* vp = vt + ((long)b * FDIM + ft * 16 + lo) * NNODE + jb + 8 * half;
            const v16bf vfrag = load16bf_split(vp);
            accS[ft] = __builtin_amdgcn_wmma_f32_16x16x32_bf16(
                           false, pfrag, false, vfrag, (short)0, accS[ft], false, false);
            accA[ft] = __builtin_amdgcn_wmma_f32_16x16x32_bf16(
                           false, afrag, false, vfrag, (short)0, accA[ft], false, false);
        }
    }

    // out = accS / l + accA   (every row has a self-loop => l > 0)
    float inv[8];
#pragma unroll
    for (int r = 0; r < 8; ++r) inv[r] = 1.0f / lrun[r];

#pragma unroll
    for (int ft = 0; ft < 8; ++ft) {
#pragma unroll
        for (int r = 0; r < 8; ++r) {
            const int m = r + 8 * half;
            out[(long)(b * NNODE + rowbase + m) * FDIM + ft * 16 + lo] =
                accS[ft][r] * inv[r] + accA[ft][r];
        }
    }
}

// ---------------------------------------------------------------------------
extern "C" void kernel_launch(void* const* d_in, const int* in_sizes, int n_in,
                              void* d_out, int out_size, void* d_ws, size_t ws_size,
                              hipStream_t stream) {
    const float* x   = (const float*)d_in[0];
    const float* adj = (const float*)d_in[1];
    const float* Wq  = (const float*)d_in[2];
    const float* Wk  = (const float*)d_in[3];
    const float* Wv  = (const float*)d_in[4];
    float* out = (float*)d_out;

    const size_t qkv_bytes = (size_t)BATCH * NNODE * FDIM * 2;  // bf16
    __bf16* qb  = (__bf16*)((char*)d_ws);
    __bf16* kbf = (__bf16*)((char*)d_ws + qkv_bytes);
    __bf16* vtb = (__bf16*)((char*)d_ws + 2 * qkv_bytes);

    dim3 blk(WAVES * 32);
    dim3 g1((BATCH * NNODE / 16) / WAVES);
    proj_kernel<<<g1, blk, 0, stream>>>(x, Wq, Wk, Wv, qb, kbf, vtb);

    dim3 g2((NNODE / 16) / WAVES, BATCH);
    attn_kernel<<<g2, blk, 0, stream>>>(adj, qb, kbf, vtb, out);
}
